// MultiHeadAttention_75393855914175
// MI455X (gfx1250) — matrix operand order repaired and checked by reading, not compiled, not run on hardware
//
#include <hip/hip_runtime.h>
#include <hip/hip_bf16.h>
#include <cstdint>

typedef __bf16 bf16_t;
typedef __attribute__((ext_vector_type(16))) __bf16 v16bf;
typedef __attribute__((ext_vector_type(8)))  __bf16 v8bf;
typedef __attribute__((ext_vector_type(8)))  float  v8f;

static constexpr int kBATCH = 2;
static constexpr int kS  = 2048;
static constexpr int kD  = 1024;
static constexpr int kH  = 16;
static constexpr int kDh = 64;
static constexpr int kM  = kBATCH * kS;   // 4096 rows

__device__ __forceinline__ bf16_t f2bf(float f) { return (bf16_t)f; }
__device__ __forceinline__ v8bf ldg8(const bf16_t* p) {
  return *reinterpret_cast<const v8bf*>(p);
}
__device__ __forceinline__ v16bf catv(v8bf a, v8bf b) {
  return __builtin_shufflevector(a, b, 0,1,2,3,4,5,6,7,8,9,10,11,12,13,14,15);
}
__device__ __forceinline__ v8f wmma_bf16(v16bf a, v16bf b, v8f c) {
  return __builtin_amdgcn_wmma_f32_16x16x32_bf16(false, a, false, b, (short)0, c,
                                                 false, false);
}

// CDNA5 async global->LDS copy (ASYNCcnt-tracked, GVS addressing mode).
__device__ __forceinline__ void async_b128(unsigned ldsOff, unsigned gOff,
                                           unsigned long long gBase) {
  asm volatile("global_load_async_to_lds_b128 %0, %1, %2"
               :: "v"(ldsOff), "v"(gOff), "s"(gBase) : "memory");
}
__device__ __forceinline__ void wait_async0() {
  asm volatile("s_wait_asynccnt 0" ::: "memory");
}
// Generic->LDS byte offset: flat LDS addressing uses addr[31:0] (ISA 10.2).
__device__ __forceinline__ unsigned lds_off(const void* p) {
  return (unsigned)(uintptr_t)p;
}

// ---------------- prep: f32 -> bf16 ----------------
__global__ void cvt_f32_bf16(const float* __restrict__ in, bf16_t* __restrict__ out,
                             int n) {
  int i = blockIdx.x * blockDim.x + threadIdx.x;
  if (i < n) out[i] = f2bf(in[i]);
}

// W [kD][kD] row-major -> Wt [kD][kD] = W^T (bf16)
__global__ void cvtT_f32_bf16(const float* __restrict__ in, bf16_t* __restrict__ out) {
  int i = blockIdx.x * blockDim.x + threadIdx.x;
  int r = i >> 10;
  int c = i & (kD - 1);
  out[c * kD + r] = f2bf(in[i]);
}

// ---------------- bf16 GEMM: C[M,N] = A[M,K] x Bt[N,K]^T ----------------
// 128x128 block tile, 4 waves, 64x64 per wave (4x4 WMMA accumulators).
// Double-buffered LDS fed by GLOBAL_LOAD_ASYNC_TO_LDS_B128.
// mode 0: f32 out, row-major [M,N]
// mode 1: bf16 out, head layout  [b, h, s, dh]   (Q, K)
// mode 2: bf16 out, transposed   [b, h, dh, s]   (V)
__global__ void __launch_bounds__(128)
gemm_bf16(const bf16_t* __restrict__ A, const bf16_t* __restrict__ Bt,
          void* __restrict__ outp, int M, int N, int K, int mode) {
  __shared__ bf16_t As[2][128 * 32];
  __shared__ bf16_t Bs[2][128 * 32];
  const int tid  = threadIdx.x;           // 0..127
  const int lane = tid & 31, wave = tid >> 5;
  const int hi = lane >> 4, l16 = lane & 15;
  const int wm = wave & 1, wn = wave >> 1;       // 2 x 2 wave grid
  const int bm = blockIdx.y * 128, bn = blockIdx.x * 128;

  v8f acc[4][4] = {};
  const int ca   = hi ? 16 : 0;          // A-frag K base (contiguous, matches B)
  const int kb16 = hi ? 16 : 0;          // B-frag K base (contiguous, matches A)

  // per-thread staging: thread t copies row t (64 bytes) of each 128x32 slab
  const unsigned aRow = (unsigned)(((bm + tid) * K) * 2);  // bytes
  const unsigned bRow = (unsigned)(((bn + tid) * K) * 2);  // bytes
  const unsigned ldsA0 = lds_off(&As[0][tid * 32]);
  const unsigned ldsB0 = lds_off(&Bs[0][tid * 32]);
  const unsigned bufB  = (unsigned)(128 * 32 * sizeof(bf16_t));  // 8192
  const unsigned long long aBase = (unsigned long long)(uintptr_t)A;
  const unsigned long long bBase = (unsigned long long)(uintptr_t)Bt;

  auto issue_slab = [&](int k0, int buf) {
    const unsigned la = ldsA0 + (unsigned)buf * bufB;
    const unsigned lb = ldsB0 + (unsigned)buf * bufB;
    const unsigned ao = aRow + (unsigned)(k0 * 2);
    const unsigned bo = bRow + (unsigned)(k0 * 2);
#pragma unroll
    for (unsigned c = 0; c < 64; c += 16) {
      async_b128(la + c, ao + c, aBase);
      async_b128(lb + c, bo + c, bBase);
    }
  };

  issue_slab(0, 0);
  int buf = 0;
  for (int k0 = 0; k0 < K; k0 += 32) {
    wait_async0();
    __syncthreads();
    if (k0 + 32 < K) issue_slab(k0 + 32, buf ^ 1);

    const bf16_t* as = As[buf];
    const bf16_t* bs = Bs[buf];
    v16bf bfrag[4];
#pragma unroll
    for (int nt = 0; nt < 4; ++nt) {
      const bf16_t* p = &bs[(wn * 64 + nt * 16 + l16) * 32];
      bfrag[nt] = catv(*(const v8bf*)(p + kb16), *(const v8bf*)(p + kb16 + 8));
    }
#pragma unroll
    for (int mt = 0; mt < 4; ++mt) {
      const bf16_t* p = &as[(wm * 64 + mt * 16 + l16) * 32];
      v16bf af = catv(*(const v8bf*)(p + ca), *(const v8bf*)(p + ca + 8));
#pragma unroll
      for (int nt = 0; nt < 4; ++nt)
        acc[mt][nt] = wmma_bf16(af, bfrag[nt], acc[mt][nt]);
    }
    buf ^= 1;
  }

  // epilogue
#pragma unroll
  for (int mt = 0; mt < 4; ++mt)
#pragma unroll
    for (int nt = 0; nt < 4; ++nt)
#pragma unroll
      for (int e = 0; e < 8; ++e) {
        int m = bm + wm * 64 + mt * 16 + e + 8 * hi;   // C layout: row = vgpr + 8*hi
        int n = bn + wn * 64 + nt * 16 + l16;          // col = lane%16
        float v = acc[mt][nt][e];
        if (mode == 0) {
          ((float*)outp)[(size_t)m * N + n] = v;
        } else {
          int bb = m >> 11, ss = m & (kS - 1);
          int hh = n >> 6,  dd = n & 63;
          bf16_t* o = (bf16_t*)outp;
          if (mode == 1)
            o[((size_t)(bb * kH + hh) * kS + ss) * kDh + dd] = f2bf(v);
          else
            o[((size_t)(bb * kH + hh) * kDh + dd) * kS + ss] = f2bf(v);
        }
      }
}

// ---------------- flash attention per head ----------------
// Q,K: [b,h,s,dh] bf16 ; Vt: [b,h,dh,s] bf16 ; ctx out: [b,s,h*dh] bf16
__global__ void __launch_bounds__(256)
attn_kernel(const bf16_t* __restrict__ Q, const bf16_t* __restrict__ Kmat,
            const bf16_t* __restrict__ Vt, bf16_t* __restrict__ ctx) {
  __shared__ bf16_t P[8][16 * 32];   // per-wave probability staging tile
  const int lane = threadIdx.x & 31, wave = threadIdx.x >> 5;
  const int hi = lane >> 4, l16 = lane & 15;
  const int g  = blockIdx.x * 8 + wave;      // global 16-query tile id
  const int qt = g & 127;
  const int h  = (g >> 7) & 15;
  const int b  = g >> 11;

  const bf16_t* Qh = Q    + (size_t)(b * kH + h) * kS * kDh;
  const bf16_t* KH = Kmat + (size_t)(b * kH + h) * kS * kDh;
  const bf16_t* VT = Vt   + (size_t)(b * kH + h) * kDh * kS;
  const int q0   = qt * 16;
  const int ca   = hi ? 16 : 0;
  const int kb16 = hi ? 16 : 0;

  // Q A-fragments for dh 0..31 and 32..63 (loaded once)
  const bf16_t* qrow = Qh + (size_t)(q0 + l16) * kDh;
  v16bf qf0 = catv(ldg8(qrow + ca),      ldg8(qrow + ca + 8));
  v16bf qf1 = catv(ldg8(qrow + 32 + ca), ldg8(qrow + 32 + ca + 8));

  v8f acc[4] = {};           // 16q x 64d accumulator (4 n-tiles)
  float mrow[8], lrow[8];
#pragma unroll
  for (int e = 0; e < 8; ++e) { mrow[e] = -1e30f; lrow[e] = 0.f; }

  for (int kt = 0; kt < kS; kt += 32) {
    // scores for two 16-key tiles, contraction over dh=64 (2 wmma each)
    v8f sc[2];
#pragma unroll
    for (int t = 0; t < 2; ++t) {
      const bf16_t* krow = KH + (size_t)(kt + t * 16 + l16) * kDh;
      v16bf kf0 = catv(ldg8(krow + kb16),      ldg8(krow + kb16 + 8));
      v16bf kf1 = catv(ldg8(krow + 32 + kb16), ldg8(krow + 32 + kb16 + 8));
      v8f z = {};
      z = wmma_bf16(qf0, kf0, z);
      sc[t] = wmma_bf16(qf1, kf1, z);
    }
    if (kt + 32 < kS)
      __builtin_prefetch(KH + (size_t)(kt + 32 + l16) * kDh, 0, 1);

    float mnew[8], al[8], v0a[8], v1a[8];
#pragma unroll
    for (int e = 0; e < 8; ++e) {
      float v0 = sc[0][e] * 0.125f; if (v0 == 0.f) v0 = -1e30f;  // reference quirk
      float v1 = sc[1][e] * 0.125f; if (v1 == 0.f) v1 = -1e30f;
      v0a[e] = v0; v1a[e] = v1;
      float rm = fmaxf(v0, v1);                    // row-wise max across 32 keys
      rm = fmaxf(rm, __shfl_xor(rm, 1, 32));
      rm = fmaxf(rm, __shfl_xor(rm, 2, 32));
      rm = fmaxf(rm, __shfl_xor(rm, 4, 32));
      rm = fmaxf(rm, __shfl_xor(rm, 8, 32));
      float mn = fmaxf(mrow[e], rm);
      al[e] = __expf(mrow[e] - mn);
      mnew[e] = mn;
    }
#pragma unroll
    for (int e = 0; e < 8; ++e) {
      float e0 = __expf(v0a[e] - mnew[e]);
      float e1 = __expf(v1a[e] - mnew[e]);
      float rs = e0 + e1;
      rs += __shfl_xor(rs, 1, 32);
      rs += __shfl_xor(rs, 2, 32);
      rs += __shfl_xor(rs, 4, 32);
      rs += __shfl_xor(rs, 8, 32);
      lrow[e] = lrow[e] * al[e] + rs;
      mrow[e] = mnew[e];
      int row = e + 8 * hi;
      P[wave][row * 32 + l16]      = f2bf(e0);   // C-layout -> row-major LDS
      P[wave][row * 32 + 16 + l16] = f2bf(e1);
#pragma unroll
      for (int nt = 0; nt < 4; ++nt) acc[nt][e] *= al[e];
    }
    asm volatile("s_wait_dscnt 0" ::: "memory");

    // re-read P as A-fragment (16q x 32k), multiply with V chunks
    const bf16_t* prow = &P[wave][l16 * 32];
    v16bf pf = catv(*(const v8bf*)(prow + ca), *(const v8bf*)(prow + ca + 8));
#pragma unroll
    for (int nt = 0; nt < 4; ++nt) {
      const bf16_t* vrow = VT + (size_t)(nt * 16 + l16) * kS + kt + kb16;
      v16bf vf = catv(ldg8(vrow), ldg8(vrow + 8));
      acc[nt] = wmma_bf16(pf, vf, acc[nt]);
    }
    asm volatile("s_wait_dscnt 0" ::: "memory");
  }

  // normalize and write context [b, s, h*dh]
#pragma unroll
  for (int nt = 0; nt < 4; ++nt)
#pragma unroll
    for (int e = 0; e < 8; ++e) {
      int row = e + 8 * hi;
      float v = acc[nt][e] / lrow[e];
      ctx[((size_t)(b * kS) + q0 + row) * kD + h * kDh + nt * 16 + l16] = f2bf(v);
    }
}

// ---------------- host launch ----------------
extern "C" void kernel_launch(void* const* d_in, const int* in_sizes, int n_in,
                              void* d_out, int out_size, void* d_ws, size_t ws_size,
                              hipStream_t stream) {
  (void)in_sizes; (void)n_in; (void)out_size; (void)ws_size;
  const float* q  = (const float*)d_in[0];
  const float* k  = (const float*)d_in[1];
  const float* v  = (const float*)d_in[2];
  const float* Wq = (const float*)d_in[3];
  const float* Wk = (const float*)d_in[4];
  const float* Wv = (const float*)d_in[5];
  const float* Wo = (const float*)d_in[6];

  char* ws = (char*)d_ws;
  const size_t ACT = (size_t)kM * kD * sizeof(bf16_t);  // 8 MB
  const size_t WT  = (size_t)kD * kD * sizeof(bf16_t);  // 2 MB
  bf16_t* Xq  = (bf16_t*)(ws + 0 * ACT);
  bf16_t* Xk  = (bf16_t*)(ws + 1 * ACT);
  bf16_t* Xv  = (bf16_t*)(ws + 2 * ACT);
  bf16_t* WqT = (bf16_t*)(ws + 3 * ACT + 0 * WT);
  bf16_t* WkT = (bf16_t*)(ws + 3 * ACT + 1 * WT);
  bf16_t* WvT = (bf16_t*)(ws + 3 * ACT + 2 * WT);
  bf16_t* WoT = (bf16_t*)(ws + 3 * ACT + 3 * WT);
  bf16_t* Qb  = (bf16_t*)(ws + 3 * ACT + 4 * WT);
  bf16_t* Kb  = (bf16_t*)(ws + 4 * ACT + 4 * WT);
  bf16_t* Vtb = (bf16_t*)(ws + 5 * ACT + 4 * WT);
  bf16_t* Ctx = (bf16_t*)(ws + 6 * ACT + 4 * WT);

  const int nAct = kM * kD;   // 4M elements
  const int nW   = kD * kD;   // 1M elements
  cvt_f32_bf16<<<nAct / 256, 256, 0, stream>>>(q, Xq, nAct);
  cvt_f32_bf16<<<nAct / 256, 256, 0, stream>>>(k, Xk, nAct);
  cvt_f32_bf16<<<nAct / 256, 256, 0, stream>>>(v, Xv, nAct);
  cvtT_f32_bf16<<<nW / 256, 256, 0, stream>>>(Wq, WqT);
  cvtT_f32_bf16<<<nW / 256, 256, 0, stream>>>(Wk, WkT);
  cvtT_f32_bf16<<<nW / 256, 256, 0, stream>>>(Wv, WvT);
  cvtT_f32_bf16<<<nW / 256, 256, 0, stream>>>(Wo, WoT);

  dim3 gg(kD / 128, kM / 128);   // (8, 32)
  gemm_bf16<<<gg, 128, 0, stream>>>(Xq, WqT, Qb,  kM, kD, kD, 1);
  gemm_bf16<<<gg, 128, 0, stream>>>(Xk, WkT, Kb,  kM, kD, kD, 1);
  gemm_bf16<<<gg, 128, 0, stream>>>(Xv, WvT, Vtb, kM, kD, kD, 2);

  const int nQTiles = kBATCH * kH * (kS / 16);   // 4096 wave-tiles
  attn_kernel<<<nQTiles / 8, 256, 0, stream>>>(Qb, Kb, Vtb, Ctx);

  gemm_bf16<<<gg, 128, 0, stream>>>(Ctx, WoT, d_out, kM, kD, kD, 0);
}
